// KNNInterpreter_27779848470992
// MI455X (gfx1250) — matrix operand order repaired and checked by reading, not compiled, not run on hardware
//
#include <hip/hip_runtime.h>
#include <stdint.h>

// ---------------------------------------------------------------------------
// 1-NN vector quantization: out[i] = argmin_n ||q_i - c_n||^2
//   q: [M=131072][C=128] f32 (x flattened), c: [K=256][C=128] f32
// Strategy: argmin_n ( ||c_n||^2 - 2 q.c_n ).  Cross term via split-bf16 WMMA
// (q=q_hi+q_lo, c=c_hi+c_lo; dot ~= hi*hi + lo*hi + hi*lo, f32 accumulate)
// giving ~fp32 accuracy at bf16 matrix-core rates.
// Codebook staged to LDS with GLOBAL_LOAD_ASYNC_TO_LDS_B128 (ASYNCcnt path).
// ---------------------------------------------------------------------------

typedef __attribute__((ext_vector_type(16))) __bf16 v16bf;
typedef __attribute__((ext_vector_type(8)))  __bf16 v8bf;
typedef __attribute__((ext_vector_type(8)))  float  v8f;

#define C_DIM 128
#define K_CB  256
#define LDS_STRIDE 136   // 128 + 8 bf16 pad -> 272B row stride, conflict-free b128 LDS loads

__device__ __forceinline__ unsigned short f32_to_bf16_rne(float v) {
    unsigned u = __float_as_uint(v);
    unsigned r = u + 0x7fffu + ((u >> 16) & 1u);
    return (unsigned short)(r >> 16);
}

// hi = truncated top-16 bits (so v - hi is exact in f32); lo = RNE(residual)
__device__ __forceinline__ void split_f32(float v, unsigned short& hi, unsigned short& lo) {
    unsigned u = __float_as_uint(v);
    hi = (unsigned short)(u >> 16);
    float hf = __uint_as_float(u & 0xffff0000u);
    lo = f32_to_bf16_rne(v - hf);
}

__device__ __forceinline__ v16bf lds_load32(const unsigned short* p) {
    v8bf a = *(const v8bf*)(const void*)(p);
    v8bf b = *(const v8bf*)(const void*)(p + 8);
    return __builtin_shufflevector(a, b, 0,1,2,3,4,5,6,7,8,9,10,11,12,13,14,15);
}

// ---------------------------------------------------------------------------
// Prep: split codebook into bf16 hi/lo planes + exact f32 squared norms.
// grid = 256 (one block per codeword), block = 128 (one thread per channel)
// ---------------------------------------------------------------------------
__global__ void __launch_bounds__(128)
prep_codebook(const float* __restrict__ cb,
              unsigned short* __restrict__ chi,
              unsigned short* __restrict__ clo,
              float* __restrict__ csq) {
    const int b = blockIdx.x;
    const int t = threadIdx.x;
    float x = cb[b * C_DIM + t];
    unsigned short h, l;
    split_f32(x, h, l);
    chi[b * C_DIM + t] = h;
    clo[b * C_DIM + t] = l;

    __shared__ float red[128];
    red[t] = x * x;
    __syncthreads();
    for (int s = 64; s > 0; s >>= 1) {
        if (t < s) red[t] += red[t + s];
        __syncthreads();
    }
    if (t == 0) csq[b] = red[0];
}

// ---------------------------------------------------------------------------
// Main kernel: 256 threads = 8 waves, each wave handles 16 queries.
// ---------------------------------------------------------------------------
__global__ void __launch_bounds__(256)
knn_argmin(const float* __restrict__ x,
           const unsigned short* __restrict__ chi_g,
           const unsigned short* __restrict__ clo_g,
           const float* __restrict__ csq_g,
           int* __restrict__ out) {
    extern __shared__ char smem_raw[];
    unsigned short* chi_s = (unsigned short*)smem_raw;
    unsigned short* clo_s = chi_s + K_CB * LDS_STRIDE;
    float*          csq_s = (float*)(clo_s + K_CB * LDS_STRIDE);

    const int tid = threadIdx.x;

    // Stage split codebook global -> LDS (padded rows) with async-to-LDS DMA.
    // GV form: vdst = LDS byte offset, v[addr] = 64-bit global address.
    // 16 iterations * 2 ops = 32 outstanding per lane (< 63 ASYNCcnt limit).
    for (int c = tid; c < (K_CB * C_DIM) / 8; c += 256) {
        int row = c >> 4, cg = c & 15;
        unsigned ldsoff_h = (unsigned)(uintptr_t)(chi_s + row * LDS_STRIDE + cg * 8);
        unsigned ldsoff_l = (unsigned)(uintptr_t)(clo_s + row * LDS_STRIDE + cg * 8);
        const void* gh = (const void*)(chi_g + c * 8);
        const void* gl = (const void*)(clo_g + c * 8);
        asm volatile("global_load_async_to_lds_b128 %0, %1, off"
                     :: "v"(ldsoff_h), "v"(gh) : "memory");
        asm volatile("global_load_async_to_lds_b128 %0, %1, off"
                     :: "v"(ldsoff_l), "v"(gl) : "memory");
    }
    if (tid < K_CB) csq_s[tid] = csq_g[tid];
    asm volatile("s_wait_asynccnt 0x0" ::: "memory");
    __syncthreads();

    const int lane   = tid & 31;
    const int wave   = tid >> 5;
    const int m      = lane & 15;       // query-in-tile (A rows / C cols mapping)
    const int hiHalf = lane >> 4;       // 0: lanes 0-15, 1: lanes 16-31
    const long qbase = ((long)blockIdx.x * 8 + wave) * 16;
    const float* qrow = x + (qbase + m) * C_DIM;
    const int s8 = hiHalf * 16;         // A-matrix K half select (16 consecutive K per half-wave)

    // Build split-bf16 A operands for all 4 k-steps (K=32 each), in this code's contiguous 16-bit
    // A layout: lane m holds K = {kb*32+s8+0..15}.
    v16bf ahi[4], alo[4];
#pragma unroll
    for (int kb = 0; kb < 4; ++kb) {
        const int kb0 = kb * 32 + s8;
        float4 f0 = *(const float4*)(qrow + kb0);
        float4 f1 = *(const float4*)(qrow + kb0 + 4);
        float4 f2 = *(const float4*)(qrow + kb0 + 8);
        float4 f3 = *(const float4*)(qrow + kb0 + 12);
        float fv[16] = {f0.x, f0.y, f0.z, f0.w, f1.x, f1.y, f1.z, f1.w,
                        f2.x, f2.y, f2.z, f2.w, f3.x, f3.y, f3.z, f3.w};
#pragma unroll
        for (int e = 0; e < 16; ++e) {
            unsigned short h, l;
            split_f32(fv[e], h, l);
            ahi[kb][e] = __builtin_bit_cast(__bf16, h);
            alo[kb][e] = __builtin_bit_cast(__bf16, l);
        }
    }

    float best[8];
    int   bidx[8];
#pragma unroll
    for (int r = 0; r < 8; ++r) { best[r] = 3.4e38f; bidx[r] = 0; }

    const int kboff = hiHalf * 16;      // B layout: 16 consecutive K per half-wave

#pragma unroll 1
    for (int t16 = 0; t16 < 16; ++t16) {
        const int n = t16 * 16 + m;     // codeword column this lane supplies/owns
        const unsigned short* pbh = chi_s + n * LDS_STRIDE + kboff;
        const unsigned short* pbl = clo_s + n * LDS_STRIDE + kboff;

        // Batch ALL 16 ds_load_b128 for this tile into distinct live registers,
        // so the compiler can use partial DScnt waits (loads drain in order)
        // instead of a full s_wait_dscnt 0 before each WMMA group.
        v16bf bhv[4], blv[4];
#pragma unroll
        for (int kb = 0; kb < 4; ++kb) {
            bhv[kb] = lds_load32(pbh + kb * 32);
            blv[kb] = lds_load32(pbl + kb * 32);
        }

        v8f acc = {};
#pragma unroll
        for (int kb = 0; kb < 4; ++kb) {
            acc = __builtin_amdgcn_wmma_f32_16x16x32_bf16(false, ahi[kb], false, bhv[kb],
                                                          (short)0, acc, false, false);
            acc = __builtin_amdgcn_wmma_f32_16x16x32_bf16(false, alo[kb], false, bhv[kb],
                                                          (short)0, acc, false, false);
            acc = __builtin_amdgcn_wmma_f32_16x16x32_bf16(false, ahi[kb], false, blv[kb],
                                                          (short)0, acc, false, false);
        }
        const float cs = csq_s[n];
#pragma unroll
        for (int r = 0; r < 8; ++r) {
            // score = ||c_n||^2 - 2 * q.c_n   (||q||^2 constant per query)
            float sc = __builtin_fmaf(-2.0f, acc[r], cs);
            if (sc < best[r]) { best[r] = sc; bidx[r] = n; }   // increasing n: keeps first min
        }
    }

    // Reduce across the 16 lanes sharing each query row (xor masks stay in half-wave).
#pragma unroll
    for (int r = 0; r < 8; ++r) {
        float bv = best[r];
        int   bi = bidx[r];
#pragma unroll
        for (int msk = 1; msk <= 8; msk <<= 1) {
            float ov = __shfl_xor(bv, msk, 32);
            int   oi = __shfl_xor(bi, msk, 32);
            if (ov < bv || (ov == bv && oi < bi)) { bv = ov; bi = oi; }
        }
        if (m == r) out[qbase + r + hiHalf * 8] = bi;
    }
}

// ---------------------------------------------------------------------------
extern "C" void kernel_launch(void* const* d_in, const int* in_sizes, int n_in,
                              void* d_out, int out_size, void* d_ws, size_t ws_size,
                              hipStream_t stream) {
    const float* x  = (const float*)d_in[0];   // [8*128*128*128] f32
    const float* cb = (const float*)d_in[1];   // [256*128] f32
    int* out = (int*)d_out;                    // [131072] int32 indices

    const int M = in_sizes[0] / C_DIM;         // 131072 queries

    // Workspace layout: chi[256*128] bf16 | clo[256*128] bf16 | csq[256] f32
    unsigned short* chi = (unsigned short*)d_ws;
    unsigned short* clo = chi + K_CB * C_DIM;
    float*          csq = (float*)(clo + K_CB * C_DIM);

    prep_codebook<<<dim3(K_CB), dim3(128), 0, stream>>>(cb, chi, clo, csq);

    const int qPerBlock = 128;                 // 8 waves * 16 queries
    size_t smem = (size_t)2 * K_CB * LDS_STRIDE * sizeof(unsigned short)
                + (size_t)K_CB * sizeof(float);
    knn_argmin<<<dim3(M / qPerBlock), dim3(256), smem, stream>>>(x, chi, clo, csq, out);
}
